// MixtralSparseMoeBlock2_2310692405614
// MI455X (gfx1250) — matrix operand order repaired and checked by reading, not compiled, not run on hardware
//
#include <hip/hip_runtime.h>
#include <hip/hip_bf16.h>

// ---------------------------------------------------------------------------
// MoE block for MI455X (gfx1250): bf16 WMMA, async-to-LDS double buffering,
// sparse token gather per expert.
// ---------------------------------------------------------------------------

#define HID 2048
#define FFN 4096
#define NEXP 8
#define NTOK 8192            // B*S = 8*1024
#define OUT_ELEMS ((size_t)NTOK * HID)

typedef __attribute__((ext_vector_type(16))) __bf16        v16bf;
typedef __attribute__((ext_vector_type(8)))  float         v8f;
typedef __attribute__((ext_vector_type(8)))  unsigned int  v8u;

__device__ __forceinline__ unsigned pack_bf2(float lo, float hi) {
  unsigned short a = __builtin_bit_cast(unsigned short, (__bf16)lo);
  unsigned short b = __builtin_bit_cast(unsigned short, (__bf16)hi);
  return (unsigned)a | ((unsigned)b << 16);
}

__device__ __forceinline__ unsigned short f2bf(float f) {
  return __builtin_bit_cast(unsigned short, (__bf16)f);
}

__device__ __forceinline__ v8f wmma_bf16(v16bf a, v16bf b, v8f c) {
  // (neg_a, A, neg_b, B, c_mod, C, reuse_a, reuse_b)
  return __builtin_amdgcn_wmma_f32_16x16x32_bf16(false, a, false, b, (short)0, c,
                                                 false, false);
}

// A-fragment (16x32 bf16, MxK): lane m = lane&15, half = lane>>4,
// VGPR j holds K pair at 2*(j&3) + 16*(j>>2) + 8*half.  LDS tile is [M][32].
__device__ __forceinline__ v16bf load_a_frag(const unsigned short* row, int half) {
  v8u au;
#pragma unroll
  for (int j = 0; j < 8; ++j) {
    int k = 2 * (j & 3) + 16 * (j >> 2) + 8 * half;
    au[j] = *(const unsigned int*)(row + k);
  }
  return __builtin_bit_cast(v16bf, au);
}

// B-fragment (32x16 bf16, KxN): lane n = lane&15, half = lane>>4,
// VGPR j holds K pair at 2*(j&3) + 16*(j>>2) + 8*half.  LDS tile is stored transposed [N][32].
__device__ __forceinline__ v16bf load_b_frag(const unsigned short* row, int half) {
  v8u bu;
#pragma unroll
  for (int j = 0; j < 8; ++j) {
    int k = 2 * (j & 3) + 16 * (j >> 2) + 8 * half;
    bu[j] = *(const unsigned int*)(row + k);
  }
  return __builtin_bit_cast(v16bf, bu);
}

// gfx1250 async copy: 16B per lane, global -> LDS, tracked by ASYNCcnt.
__device__ __forceinline__ void async_copy_b128(unsigned ldsAddr, const void* g) {
  unsigned long long ga = (unsigned long long)g;
  asm volatile("global_load_async_to_lds_b128 %0, %1, off"
               :: "v"(ldsAddr), "v"(ga) : "memory");
}
__device__ __forceinline__ void wait_async0() {
  asm volatile("s_wait_asynccnt 0" ::: "memory");
}

// ---------------------------------------------------------------------------
// Kernel 0: zero the output accumulator region + per-expert counters.
// ---------------------------------------------------------------------------
__global__ __launch_bounds__(256) void zero_out_kernel(float* __restrict__ out,
                                                       int* __restrict__ counts) {
  size_t i = (size_t)blockIdx.x * blockDim.x + threadIdx.x;   // one float4 each
  float4 z = {0.f, 0.f, 0.f, 0.f};
  ((float4*)out)[i] = z;
  if (blockIdx.x == 0 && threadIdx.x < NEXP) counts[threadIdx.x] = 0;
}

// ---------------------------------------------------------------------------
// Kernel 1: X fp32 -> bf16 once (activations then reused at bf16).
// ---------------------------------------------------------------------------
__global__ __launch_bounds__(256) void convert_x_kernel(const float* __restrict__ x,
                                                        unsigned short* __restrict__ xbf) {
  size_t i = (size_t)blockIdx.x * blockDim.x + threadIdx.x;   // one float4 each
  float4 v = ((const float4*)x)[i];
  uint2 o;
  o.x = pack_bf2(v.x, v.y);
  o.y = pack_bf2(v.z, v.w);
  ((uint2*)xbf)[i] = o;
}

// ---------------------------------------------------------------------------
// Kernel 2: router.  One wave32 per token: coalesced fp32 dot with gate_w,
// butterfly reduce, softmax + top-2, emit raw logits + per-expert gather lists.
// ---------------------------------------------------------------------------
__global__ __launch_bounds__(256) void router_kernel(const float* __restrict__ x,
                                                     const float* __restrict__ gw,
                                                     float* __restrict__ logits_out,
                                                     int* __restrict__ idx,
                                                     float* __restrict__ wgt,
                                                     int* __restrict__ counts) {
  const int lane = threadIdx.x & 31;
  const int wave = threadIdx.x >> 5;
  const int t = blockIdx.x * 8 + wave;

  float acc[NEXP] = {0.f, 0.f, 0.f, 0.f, 0.f, 0.f, 0.f, 0.f};
  const float* xr = x + (size_t)t * HID;
  for (int i = lane; i < HID; i += 32) {
    float xv = xr[i];
    const float* g = gw + (size_t)i * NEXP;
#pragma unroll
    for (int e = 0; e < NEXP; ++e) acc[e] += xv * g[e];
  }
#pragma unroll
  for (int off = 16; off > 0; off >>= 1) {
#pragma unroll
    for (int e = 0; e < NEXP; ++e) acc[e] += __shfl_xor(acc[e], off, 32);
  }

  if (lane == 0) {
#pragma unroll
    for (int e = 0; e < NEXP; ++e) logits_out[(size_t)t * NEXP + e] = acc[e];
    float mx = acc[0];
#pragma unroll
    for (int e = 1; e < NEXP; ++e) mx = fmaxf(mx, acc[e]);
    float p[NEXP];
#pragma unroll
    for (int e = 0; e < NEXP; ++e) p[e] = __expf(acc[e] - mx);
    // top-2 (ties -> lower index, matching lax.top_k)
    int i1 = 0;
#pragma unroll
    for (int e = 1; e < NEXP; ++e) if (p[e] > p[i1]) i1 = e;
    int i2 = (i1 == 0) ? 1 : 0;
#pragma unroll
    for (int e = 0; e < NEXP; ++e) if (e != i1 && p[e] > p[i2]) i2 = e;
    float s = p[i1] + p[i2];
    float w1v = p[i1] / s, w2v = p[i2] / s;
    int p1 = atomicAdd(&counts[i1], 1);
    idx[(size_t)i1 * NTOK + p1] = t;
    wgt[(size_t)i1 * NTOK + p1] = w1v;
    int p2 = atomicAdd(&counts[i2], 1);
    idx[(size_t)i2 * NTOK + p2] = t;
    wgt[(size_t)i2 * NTOK + p2] = w2v;
  }
}

// ---------------------------------------------------------------------------
// Kernel 3: gate+up projection for one expert's token list.
// Block tile 64 tok x 128 ffn, K in steps of 32, double-buffered LDS:
//   A tile via global_load_async_to_lds_b128, B tiles via regs + packed b32
//   stores (fp32 -> bf16 convert while staging).  8 waves, each 2x2 WMMA tiles
//   for both gate and up.
// ---------------------------------------------------------------------------
__global__ __launch_bounds__(256) void gateup_kernel(
    const unsigned short* __restrict__ xbf,   // [NTOK, HID] bf16
    const float* __restrict__ w1e,            // [HID, FFN] fp32 (expert slice)
    const float* __restrict__ w3e,            // [HID, FFN]
    const int* __restrict__ idxe,             // [NTOK] gather list
    const int* __restrict__ cntp,             // expert token count
    unsigned short* __restrict__ H) {         // [NTOK, FFN] bf16 (list order)
  const int cnt = *cntp;
  const int m0 = blockIdx.x * 64;
  if (m0 >= cnt) return;                      // uniform early-exit
  const int f0 = blockIdx.y * 128;

  __shared__ int s_tid[64];
  __shared__ __attribute__((aligned(16))) unsigned short sA[2][64][32];
  __shared__ __attribute__((aligned(16))) unsigned short sB1[2][128][32];
  __shared__ __attribute__((aligned(16))) unsigned short sB3[2][128][32];

  const int tid = threadIdx.x;
  if (tid < 64) {
    int r = m0 + tid;
    s_tid[tid] = (r < cnt) ? idxe[r] : 0;     // clamp pad rows to token 0
  }
  __syncthreads();

  // --- staging assignments ---
  const int arow = tid >> 2, acol = (tid & 3) * 8;     // A: 64 x 32, uint4/lane
  const int kp = tid >> 4;                              // B k-pair 0..15
  const int fb = (tid & 15) * 8;                        // B f-group 0..120

  const unsigned short* aGlob = xbf + (size_t)s_tid[arow] * HID + acol;
  const unsigned aLds[2] = { (unsigned)(size_t)&sA[0][arow][acol],
                             (unsigned)(size_t)&sA[1][arow][acol] };
  const float* b1Glob = w1e + (size_t)(2 * kp) * FFN + f0 + fb;
  const float* b3Glob = w3e + (size_t)(2 * kp) * FFN + f0 + fb;

  const int lane = tid & 31, wave = tid >> 5;
  const int wm = wave >> 2, wn = wave & 3;    // 2x4 wave grid
  const int half = lane >> 4, nl = lane & 15;

  v8f vz = {0.f, 0.f, 0.f, 0.f, 0.f, 0.f, 0.f, 0.f};
  v8f cg[2][2], cu[2][2];
#pragma unroll
  for (int i = 0; i < 2; ++i)
#pragma unroll
    for (int j = 0; j < 2; ++j) { cg[i][j] = vz; cu[i][j] = vz; }

  float4 rb[8];   // staged B regs: [0..3]=w1 {k,f0-3},{k,f4-7},{k+1,f0-3},{k+1,f4-7}; [4..7]=w3
  auto loadB = [&](int step) {
    const float* p1 = b1Glob + (size_t)step * 32 * FFN;
    const float* p3 = b3Glob + (size_t)step * 32 * FFN;
    rb[0] = *(const float4*)(p1);
    rb[1] = *(const float4*)(p1 + 4);
    rb[2] = *(const float4*)(p1 + FFN);
    rb[3] = *(const float4*)(p1 + FFN + 4);
    rb[4] = *(const float4*)(p3);
    rb[5] = *(const float4*)(p3 + 4);
    rb[6] = *(const float4*)(p3 + FFN);
    rb[7] = *(const float4*)(p3 + FFN + 4);
  };
  auto storeB = [&](int buf) {
    unsigned* d1 = (unsigned*)&sB1[buf][fb][2 * kp];
    unsigned* d3 = (unsigned*)&sB3[buf][fb][2 * kp];
    const int strd = 16;                      // 32 ushorts / 2 per row
    d1[0 * strd] = pack_bf2(rb[0].x, rb[2].x);
    d1[1 * strd] = pack_bf2(rb[0].y, rb[2].y);
    d1[2 * strd] = pack_bf2(rb[0].z, rb[2].z);
    d1[3 * strd] = pack_bf2(rb[0].w, rb[2].w);
    d1[4 * strd] = pack_bf2(rb[1].x, rb[3].x);
    d1[5 * strd] = pack_bf2(rb[1].y, rb[3].y);
    d1[6 * strd] = pack_bf2(rb[1].z, rb[3].z);
    d1[7 * strd] = pack_bf2(rb[1].w, rb[3].w);
    d3[0 * strd] = pack_bf2(rb[4].x, rb[6].x);
    d3[1 * strd] = pack_bf2(rb[4].y, rb[6].y);
    d3[2 * strd] = pack_bf2(rb[4].z, rb[6].z);
    d3[3 * strd] = pack_bf2(rb[4].w, rb[6].w);
    d3[4 * strd] = pack_bf2(rb[5].x, rb[7].x);
    d3[5 * strd] = pack_bf2(rb[5].y, rb[7].y);
    d3[6 * strd] = pack_bf2(rb[5].z, rb[7].z);
    d3[7 * strd] = pack_bf2(rb[5].w, rb[7].w);
  };

  constexpr int NSTEP = HID / 32;
  // prologue: fill buffer 0
  async_copy_b128(aLds[0], aGlob);
  loadB(0);
  storeB(0);
  wait_async0();
  __syncthreads();

  for (int i = 0; i < NSTEP; ++i) {
    const int cur = i & 1, nxt = cur ^ 1;
    const bool more = (i + 1) < NSTEP;
    if (more) {
      async_copy_b128(aLds[nxt], aGlob + (size_t)(i + 1) * 32);
      loadB(i + 1);
      if (i + 2 < NSTEP) {                    // stream-ahead hint
        __builtin_prefetch(b1Glob + (size_t)(i + 2) * 32 * FFN, 0, 1);
        __builtin_prefetch(b3Glob + (size_t)(i + 2) * 32 * FFN, 0, 1);
      }
    }

    v16bf a[2], b1[2], b3[2];
#pragma unroll
    for (int mt = 0; mt < 2; ++mt)
      a[mt] = load_a_frag(&sA[cur][wm * 32 + mt * 16 + nl][0], half);
#pragma unroll
    for (int nt = 0; nt < 2; ++nt) {
      b1[nt] = load_b_frag(&sB1[cur][wn * 32 + nt * 16 + nl][0], half);
      b3[nt] = load_b_frag(&sB3[cur][wn * 32 + nt * 16 + nl][0], half);
    }
#pragma unroll
    for (int mt = 0; mt < 2; ++mt)
#pragma unroll
      for (int nt = 0; nt < 2; ++nt) {
        cg[mt][nt] = wmma_bf16(a[mt], b1[nt], cg[mt][nt]);
        cu[mt][nt] = wmma_bf16(a[mt], b3[nt], cu[mt][nt]);
      }

    if (more) {
      storeB(nxt);
      wait_async0();
    }
    __syncthreads();
  }

  // ---- epilogue: silu(gate) * up -> bf16 H in list order ----
#pragma unroll
  for (int mt = 0; mt < 2; ++mt)
#pragma unroll
    for (int nt = 0; nt < 2; ++nt) {
      int col = f0 + wn * 32 + nt * 16 + nl;
#pragma unroll
      for (int r = 0; r < 8; ++r) {
        float g = cg[mt][nt][r];
        float h = (g / (1.f + __expf(-g))) * cu[mt][nt][r];
        int row = m0 + wm * 32 + mt * 16 + 8 * half + r;
        H[(size_t)row * FFN + col] = f2bf(h);
      }
    }
}

// ---------------------------------------------------------------------------
// Kernel 4: down projection + weighted scatter-accumulate into output.
// Block tile 64 tok x 128 hid, K = FFN in steps of 32, same pipeline.
// ---------------------------------------------------------------------------
__global__ __launch_bounds__(256) void down_kernel(
    const unsigned short* __restrict__ H,     // [NTOK, FFN] bf16 (list order)
    const float* __restrict__ w2e,            // [FFN, HID] fp32 (expert slice)
    const int* __restrict__ idxe,
    const float* __restrict__ wgte,
    const int* __restrict__ cntp,
    float* __restrict__ out) {                // [NTOK, HID] fp32
  const int cnt = *cntp;
  const int m0 = blockIdx.x * 64;
  if (m0 >= cnt) return;
  const int h0 = blockIdx.y * 128;

  __shared__ int s_tid[64];
  __shared__ float s_w[64];
  __shared__ __attribute__((aligned(16))) unsigned short sA[2][64][32];
  __shared__ __attribute__((aligned(16))) unsigned short sB[2][128][32];

  const int tid = threadIdx.x;
  if (tid < 64) {
    int r = m0 + tid;
    bool v = r < cnt;
    s_tid[tid] = v ? idxe[r] : 0;
    s_w[tid] = v ? wgte[r] : 0.f;
  }
  __syncthreads();

  const int arow = tid >> 2, acol = (tid & 3) * 8;
  const int kp = tid >> 4;
  const int fb = (tid & 15) * 8;

  const unsigned short* aGlob = H + (size_t)(m0 + arow) * FFN + acol;
  const unsigned aLds[2] = { (unsigned)(size_t)&sA[0][arow][acol],
                             (unsigned)(size_t)&sA[1][arow][acol] };
  const float* bGlob = w2e + (size_t)(2 * kp) * HID + h0 + fb;

  const int lane = tid & 31, wave = tid >> 5;
  const int wm = wave >> 2, wn = wave & 3;
  const int half = lane >> 4, nl = lane & 15;

  v8f vz = {0.f, 0.f, 0.f, 0.f, 0.f, 0.f, 0.f, 0.f};
  v8f c[2][2];
#pragma unroll
  for (int i = 0; i < 2; ++i)
#pragma unroll
    for (int j = 0; j < 2; ++j) c[i][j] = vz;

  float4 rb[4];
  auto loadB = [&](int step) {
    const float* p = bGlob + (size_t)step * 32 * HID;
    rb[0] = *(const float4*)(p);
    rb[1] = *(const float4*)(p + 4);
    rb[2] = *(const float4*)(p + HID);
    rb[3] = *(const float4*)(p + HID + 4);
  };
  auto storeB = [&](int buf) {
    unsigned* d = (unsigned*)&sB[buf][fb][2 * kp];
    const int strd = 16;
    d[0 * strd] = pack_bf2(rb[0].x, rb[2].x);
    d[1 * strd] = pack_bf2(rb[0].y, rb[2].y);
    d[2 * strd] = pack_bf2(rb[0].z, rb[2].z);
    d[3 * strd] = pack_bf2(rb[0].w, rb[2].w);
    d[4 * strd] = pack_bf2(rb[1].x, rb[3].x);
    d[5 * strd] = pack_bf2(rb[1].y, rb[3].y);
    d[6 * strd] = pack_bf2(rb[1].z, rb[3].z);
    d[7 * strd] = pack_bf2(rb[1].w, rb[3].w);
  };

  constexpr int NSTEP = FFN / 32;
  async_copy_b128(aLds[0], aGlob);
  loadB(0);
  storeB(0);
  wait_async0();
  __syncthreads();

  for (int i = 0; i < NSTEP; ++i) {
    const int cur = i & 1, nxt = cur ^ 1;
    const bool more = (i + 1) < NSTEP;
    if (more) {
      async_copy_b128(aLds[nxt], aGlob + (size_t)(i + 1) * 32);
      loadB(i + 1);
      if (i + 2 < NSTEP)
        __builtin_prefetch(bGlob + (size_t)(i + 2) * 32 * HID, 0, 1);
    }

    v16bf a[2], b[2];
#pragma unroll
    for (int mt = 0; mt < 2; ++mt)
      a[mt] = load_a_frag(&sA[cur][wm * 32 + mt * 16 + nl][0], half);
#pragma unroll
    for (int nt = 0; nt < 2; ++nt)
      b[nt] = load_b_frag(&sB[cur][wn * 32 + nt * 16 + nl][0], half);
#pragma unroll
    for (int mt = 0; mt < 2; ++mt)
#pragma unroll
      for (int nt = 0; nt < 2; ++nt)
        c[mt][nt] = wmma_bf16(a[mt], b[nt], c[mt][nt]);

    if (more) {
      storeB(nxt);
      wait_async0();
    }
    __syncthreads();
  }

  // weighted scatter accumulate (disjoint tokens within a launch -> no atomics)
#pragma unroll
  for (int mt = 0; mt < 2; ++mt)
#pragma unroll
    for (int nt = 0; nt < 2; ++nt) {
      int col = h0 + wn * 32 + nt * 16 + nl;
#pragma unroll
      for (int r = 0; r < 8; ++r) {
        int lrow = wm * 32 + mt * 16 + 8 * half + r;
        if (m0 + lrow < cnt) {
          size_t o = (size_t)s_tid[lrow] * HID + col;
          out[o] += s_w[lrow] * c[mt][nt][r];
        }
      }
    }
}

// ---------------------------------------------------------------------------
// Host launcher
// ---------------------------------------------------------------------------
extern "C" void kernel_launch(void* const* d_in, const int* in_sizes, int n_in,
                              void* d_out, int out_size, void* d_ws, size_t ws_size,
                              hipStream_t stream) {
  const float* x  = (const float*)d_in[0];   // [8,1024,2048]
  const float* gw = (const float*)d_in[1];   // [2048,8]
  const float* w1 = (const float*)d_in[2];   // [8,2048,4096]
  const float* w3 = (const float*)d_in[3];   // [8,2048,4096]
  const float* w2 = (const float*)d_in[4];   // [8,4096,2048]

  float* out    = (float*)d_out;
  float* logits = out + OUT_ELEMS;           // [8192, 8]

  char* ws = (char*)d_ws;
  const size_t XBF_BYTES = (size_t)NTOK * HID * 2;
  const size_t H_BYTES   = (size_t)NTOK * FFN * 2;
  const size_t IDX_BYTES = (size_t)NEXP * NTOK * 4;
  unsigned short* xbf = (unsigned short*)ws;
  unsigned short* H   = (unsigned short*)(ws + XBF_BYTES);
  int*   idx    = (int*)(ws + XBF_BYTES + H_BYTES);
  float* wgt    = (float*)(ws + XBF_BYTES + H_BYTES + IDX_BYTES);
  int*   counts = (int*)(ws + XBF_BYTES + H_BYTES + 2 * IDX_BYTES);

  zero_out_kernel<<<16384, 256, 0, stream>>>(out, counts);
  convert_x_kernel<<<16384, 256, 0, stream>>>(x, xbf);
  router_kernel<<<1024, 256, 0, stream>>>(x, gw, logits, idx, wgt, counts);

  for (int e = 0; e < NEXP; ++e) {
    gateup_kernel<<<dim3(NTOK / 64, FFN / 128), 256, 0, stream>>>(
        xbf, w1 + (size_t)e * HID * FFN, w3 + (size_t)e * HID * FFN,
        idx + (size_t)e * NTOK, counts + e, H);
    down_kernel<<<dim3(NTOK / 64, HID / 128), 256, 0, stream>>>(
        H, w2 + (size_t)e * FFN * HID,
        idx + (size_t)e * NTOK, wgt + (size_t)e * NTOK, counts + e, out);
  }
}
